// _RefAttention_25159918420617
// MI455X (gfx1250) — compile-verified, matrix operand order checked by reading, not run on hardware
//
#include <hip/hip_runtime.h>

// ---------------------------------------------------------------------------
// Types
// ---------------------------------------------------------------------------
typedef __attribute__((ext_vector_type(16))) __bf16 bf16x16;
typedef __attribute__((ext_vector_type(8)))  float  f32x8;
typedef __attribute__((ext_vector_type(4)))  unsigned int u32x4;
typedef __attribute__((ext_vector_type(8)))  int i32x8;
typedef __attribute__((ext_vector_type(4)))  int i32x4;
typedef unsigned short ushort_t;
typedef unsigned long long u64_t;

// Tensor Data Mover availability (device pass only; host pass sees false)
#if defined(__AMDGCN__) && __has_builtin(__builtin_amdgcn_tensor_load_to_lds)
#define HAS_TDM 1
#else
#define HAS_TDM 0
#endif

// round-to-nearest-even f32 -> bf16 (bit pattern as ushort)
__device__ __forceinline__ ushort_t f2bf(float f) {
  unsigned u = __float_as_uint(f);
  u += 0x7fffu + ((u >> 16) & 1u);
  return (ushort_t)(u >> 16);
}

// D = A*B + C on the bf16 WMMA pipe, f32 accumulation.
__device__ __forceinline__ f32x8 wmma_bf16(bf16x16 a, bf16x16 b, f32x8 c) {
  return __builtin_amdgcn_wmma_f32_16x16x32_bf16(
      /*neg_a=*/false, a, /*neg_b=*/false, b,
      /*c_mod=*/(short)0, c, /*reuse_a=*/false, /*reuse_b=*/false);
}

// Load a 16x32 A-fragment (or 32x16 B-fragment from a row-major [n][k] panel)
// from LDS.  ISA 7.12.2 16-bit A layout: lanes 0-15 hold row M=lane with
// K=0..7 in VGPR0-3 and K=16..23 in VGPR4-7; lanes 16-31 hold K=8..15 and
// K=24..31.  Two 16B LDS reads per lane reproduce that exactly.
__device__ __forceinline__ bf16x16 load_frag(const ushort_t* base, int row0,
                                             int k0, int ld, int lane) {
  const int half = (lane >> 4) & 1;
  const int r    = lane & 15;
  const ushort_t* p = base + (row0 + r) * ld + k0 + half * 8;
  union { bf16x16 v; u32x4 u[2]; } f;
  f.u[0] = *(const u32x4*)(p);
  f.u[1] = *(const u32x4*)(p + 16);
  return f.v;
}

__device__ __forceinline__ void tensor_wait0() {
#if defined(__AMDGCN__) && __has_builtin(__builtin_amdgcn_s_wait_tensorcnt)
  __builtin_amdgcn_s_wait_tensorcnt(0);
#elif defined(__AMDGCN__)
  asm volatile("s_wait_tensorcnt 0x0" ::: "memory");
#endif
}

#if HAS_TDM
// Issue a TDM 2-D tile load (bf16 elements) via a hand-packed D#.
// D# group0: count=1 | lds_addr | global_addr | type=2 ("image").
// D# group1: data_size=2B, tensor_dim0/1 == tile_dim0/1 (no OOB), row stride.
// One wave issues this for the whole workgroup; completion via TENSORcnt.
__device__ __forceinline__ void tdm_load_2d_bf16(unsigned lds_byte_addr,
                                                 u64_t gaddr,
                                                 unsigned tile_d0,   // elems/row
                                                 unsigned tile_d1,   // rows
                                                 u64_t row_stride) { // elems
  u32x4 g0;
  g0[0] = 1u;                                            // count=1, user mode
  g0[1] = lds_byte_addr;                                 // bits [63:32]
  g0[2] = (unsigned)(gaddr & 0xffffffffu);               // bits [95:64]
  g0[3] = (unsigned)((gaddr >> 32) & 0x01ffffffu)        // bits [120:96]
        | (2u << 30);                                    // type=2
  i32x8 g1;
  g1[0] = (int)(1u << 16);                               // data_size=2B
  g1[1] = (int)((tile_d0 & 0xffffu) << 16);              // tensor_dim0 lo16
  g1[2] = (int)(((tile_d0 >> 16) & 0xffffu) |            // tensor_dim0 hi16
                ((tile_d1 & 0xffffu) << 16));            // tensor_dim1 lo16
  g1[3] = (int)(((tile_d1 >> 16) & 0xffffu) |            // tensor_dim1 hi16
                ((tile_d0 & 0xffffu) << 16));            // tile_dim0
  g1[4] = (int)(tile_d1 & 0xffffu);                      // tile_dim1, tile_dim2=0
  g1[5] = (int)(unsigned)(row_stride & 0xffffffffu);     // dim0_stride lo32
  g1[6] = (int)(unsigned)((row_stride >> 32) & 0xffffu); // dim0_stride hi16
  g1[7] = 0;
  i32x4 z4 = {0, 0, 0, 0};
#if defined(__clang_major__) && (__clang_major__ >= 23)
  i32x8 z8 = {0, 0, 0, 0, 0, 0, 0, 0};
  __builtin_amdgcn_tensor_load_to_lds(g0, g1, z4, z4, z8, 0);
#else
  __builtin_amdgcn_tensor_load_to_lds(g0, g1, z4, z4, 0);
#endif
}

__device__ __forceinline__ unsigned lds_off(const void* p) {
  return (unsigned)(u64_t)p;   // flat LDS addressing uses addr[31:0]
}
#endif  // HAS_TDM

// ---------------------------------------------------------------------------
// f32 -> bf16 conversion
// ---------------------------------------------------------------------------
__global__ void cvt_bf16_kernel(const float* __restrict__ in,
                                ushort_t* __restrict__ out, size_t n) {
  size_t i = (size_t)blockIdx.x * blockDim.x + threadIdx.x;
  if (i < n) out[i] = f2bf(in[i]);
}

// ---------------------------------------------------------------------------
// Y[M,N] = X[M,K] * W[N,K]^T   (nn.Linear), bf16 inputs, f32 or bf16 output.
// Block tile 128x64, 8 waves, each wave owns a 32x32 patch (2x2 WMMA tiles).
// ---------------------------------------------------------------------------
__global__ __launch_bounds__(256)
void gemm_xwT_wmma(const ushort_t* __restrict__ A, const ushort_t* __restrict__ W,
                   float* __restrict__ Cf, ushort_t* __restrict__ Cb,
                   int N, int K) {
  __shared__ ushort_t As[128 * 32];   // 8 KB
  __shared__ ushort_t Ws[64 * 32];    // 4 KB
  const int tid  = threadIdx.x;
  const int lane = tid & 31;
  const int wave = tid >> 5;
  const int m0 = blockIdx.y * 128;
  const int n0 = blockIdx.x * 64;
  const int wm = wave & 3;    // 4 waves along M (32 rows each)
  const int wn = wave >> 2;   // 2 waves along N (32 cols each)

  f32x8 acc[2][2] = {};

  for (int k0 = 0; k0 < K; k0 += 32) {
    {   // stage A tile 128x32 (16 bf16 per thread)
      int row = tid >> 1;
      int col = (tid & 1) * 16;
      const u32x4* src = (const u32x4*)(A + (size_t)(m0 + row) * K + k0 + col);
      u32x4* dst = (u32x4*)(As + row * 32 + col);
      dst[0] = src[0];
      dst[1] = src[1];
    }
    {   // stage W tile 64x32 (8 bf16 per thread)
      int row = tid >> 2;
      int col = (tid & 3) * 8;
      *(u32x4*)(Ws + row * 32 + col) =
          *(const u32x4*)(W + (size_t)(n0 + row) * K + k0 + col);
    }
    // warm L2/WGP$ for the next k-tile (global_prefetch_b8)
    if (k0 + 32 < K) {
      __builtin_prefetch(A + (size_t)(m0 + (tid >> 1)) * K + k0 + 32, 0, 1);
      if (tid < 64)
        __builtin_prefetch(W + (size_t)(n0 + tid) * K + k0 + 32, 0, 1);
    }
    __syncthreads();
    bf16x16 a0 = load_frag(As, wm * 32 + 0,  0, 32, lane);
    bf16x16 a1 = load_frag(As, wm * 32 + 16, 0, 32, lane);
    bf16x16 b0 = load_frag(Ws, wn * 32 + 0,  0, 32, lane);
    bf16x16 b1 = load_frag(Ws, wn * 32 + 16, 0, 32, lane);
    acc[0][0] = wmma_bf16(a0, b0, acc[0][0]);
    acc[0][1] = wmma_bf16(a0, b1, acc[0][1]);
    acc[1][0] = wmma_bf16(a1, b0, acc[1][0]);
    acc[1][1] = wmma_bf16(a1, b1, acc[1][1]);
    __syncthreads();
  }

  // C/D layout: VGPR e -> row e (+8 for lanes 16-31), col = lane&15
  const int half = (lane >> 4) & 1;
  const int cl   = lane & 15;
  for (int i = 0; i < 2; ++i)
    for (int j = 0; j < 2; ++j)
      for (int e = 0; e < 8; ++e) {
        int row = m0 + wm * 32 + i * 16 + e + half * 8;
        int col = n0 + wn * 32 + j * 16 + cl;
        float v = acc[i][j][e];
        if (Cf) Cf[(size_t)row * N + col] = v;
        else    Cb[(size_t)row * N + col] = f2bf(v);
      }
}

// ---------------------------------------------------------------------------
// Per-head RMSNorm (+ optional weight) + optional RoPE, transposing
// [S, NH*D] f32 -> [NH, S, D] bf16.  One block per (s, h), D=128 threads.
// ---------------------------------------------------------------------------
__global__ __launch_bounds__(128)
void norm_rope_kernel(const float* __restrict__ X, const float* __restrict__ w,
                      const float* __restrict__ cosT, const float* __restrict__ sinT,
                      ushort_t* __restrict__ out, int NH, int S, int do_rope) {
  __shared__ float red[128];
  __shared__ float xs[128];
  const int d = threadIdx.x;
  const int h = blockIdx.x % NH;
  const int s = blockIdx.x / NH;
  float x = X[(size_t)s * (NH * 128) + h * 128 + d];
  red[d] = x * x;
  __syncthreads();
  for (int off = 64; off > 0; off >>= 1) {
    if (d < off) red[d] += red[d + off];
    __syncthreads();
  }
  float inv = rsqrtf(red[0] * (1.0f / 128.0f) + 1e-6f);
  float n = x * inv * (w ? w[d] : 1.0f);
  xs[d] = n;
  __syncthreads();
  float o = n;
  if (do_rope) {
    float rot = (d < 64) ? -xs[d + 64] : xs[d - 64];
    o = n * cosT[s * 128 + d] + rot * sinT[s * 128 + d];
  }
  out[((size_t)h * S + s) * 128 + d] = f2bf(o);
}

// ---------------------------------------------------------------------------
// Causal GQA flash attention (scale = 1.0, matching the reference).
// Block = 256 threads (8 waves) handles 32 queries x D=128 for one head.
// Key tiles of 64 positions; only key blocks <= query block are visited.
// Q and K tiles are staged by the Tensor Data Mover when available
// (one TDM descriptor per tile, TENSORcnt-tracked); V is staged transposed
// by the vector pipe (TDM cannot transpose).
// ---------------------------------------------------------------------------
__global__ __launch_bounds__(256)
void flash_attn_kernel(const ushort_t* __restrict__ Q,   // [H][S][128]
                       const ushort_t* __restrict__ Kx,  // [KV][S][128]
                       const ushort_t* __restrict__ Vx,  // [KV][S][128]
                       ushort_t* __restrict__ O,         // [S][H*128] bf16
                       int S, int H, int KV) {
  __shared__ ushort_t Qs[32 * 128];   // 8 KB
  __shared__ ushort_t Ks[64 * 128];   // 16 KB
  __shared__ ushort_t Vts[128 * 64];  // 16 KB  V^T: [d][key]
  __shared__ float    Ss[32 * 64];    // 8 KB
  __shared__ ushort_t Ps[32 * 64];    // 4 KB
  __shared__ float mrow[32], lrow[32], crow[32];

  const int tid  = threadIdx.x;
  const int lane = tid & 31;
  const int wave = tid >> 5;
  const int rb = blockIdx.x;          // 32-query block
  const int h  = blockIdx.y;
  const int kv = h / (H / KV);        // GQA group
  const int q0 = rb * 32;

  const ushort_t* Qh = Q  + (size_t)h  * S * 128;
  const ushort_t* Kh = Kx + (size_t)kv * S * 128;
  const ushort_t* Vh = Vx + (size_t)kv * S * 128;

#if HAS_TDM
  if (wave == 0)   // one DMA descriptor stages the whole 32x128 Q tile
    tdm_load_2d_bf16(lds_off(Qs), (u64_t)(const void*)(Qh + (size_t)q0 * 128),
                     128, 32, 128);
#else
  {   // stage Q tile 32x128
    int row = tid >> 3;
    int col = (tid & 7) * 16;
    const u32x4* src = (const u32x4*)(Qh + (size_t)(q0 + row) * 128 + col);
    u32x4* dst = (u32x4*)(Qs + row * 128 + col);
    dst[0] = src[0];
    dst[1] = src[1];
  }
#endif
  if (tid < 32) { mrow[tid] = -1e30f; lrow[tid] = 0.0f; }

  const int sm  = wave & 1;          // score / O row-tile (16 rows)
  const int sn  = wave >> 1;         // score col-tile (0..3)
  const int dt0 = sn * 2;            // O d-tile base (covers 8 d-tiles)
  f32x8 oacc[2] = {};

  const int nkb = rb / 2 + 1;        // causal: key blocks up to the diagonal
  for (int kbi = 0; kbi < nkb; ++kbi) {
    const int key0 = kbi * 64;
#if HAS_TDM
    if (wave == 0)   // TDM stages the 64x128 K tile for the whole workgroup
      tdm_load_2d_bf16(lds_off(Ks), (u64_t)(const void*)(Kh + (size_t)key0 * 128),
                       128, 64, 128);
#else
    {   // stage K tile 64x128 row-major (B-fragment friendly for Q*K^T)
      int row = tid >> 2;
      int colc = (tid & 3) * 32;
      for (int c = 0; c < 4; ++c)
        *(u32x4*)(Ks + row * 128 + colc + c * 8) =
            *(const u32x4*)(Kh + (size_t)(key0 + row) * 128 + colc + c * 8);
    }
#endif
    {   // stage V tile transposed: Vts[d][key]  (B-fragment friendly for P*V)
      int row = tid >> 2;
      int colc = (tid & 3) * 32;
      for (int c = 0; c < 4; ++c) {
        u32x4 v = *(const u32x4*)(Vh + (size_t)(key0 + row) * 128 + colc + c * 8);
        const ushort_t* e = (const ushort_t*)&v;
        for (int t = 0; t < 8; ++t)
          Vts[(colc + c * 8 + t) * 64 + row] = e[t];
      }
    }
    // warm caches for the next key tile (global_prefetch_b8)
    if (kbi + 1 < nkb) {
      __builtin_prefetch(Kh + (size_t)(key0 + 64) * 128 + tid * 32, 0, 1);
      __builtin_prefetch(Vh + (size_t)(key0 + 64) * 128 + tid * 32, 0, 1);
    }
#if HAS_TDM
    if (wave == 0) tensor_wait0();   // TENSORcnt==0 before LDS is consumed
#endif
    __syncthreads();

    // ---- scores: S = Q * K^T (each wave one 16x16 tile, K-loop over D) ----
    f32x8 sc = {};
    for (int kk = 0; kk < 4; ++kk) {
      bf16x16 aq = load_frag(Qs, sm * 16, kk * 32, 128, lane);
      bf16x16 bk = load_frag(Ks, sn * 16, kk * 32, 128, lane);
      sc = wmma_bf16(aq, bk, sc);
    }
    {
      const int half = lane >> 4, cl = lane & 15;
      for (int e = 0; e < 8; ++e)
        Ss[(sm * 16 + e + half * 8) * 64 + sn * 16 + cl] = sc[e];
    }
    __syncthreads();

    // ---- online softmax: one thread per query row, f32 stats ----
    if (tid < 32) {
      const int r = tid;
      const int qa = q0 + r;
      float m_old = mrow[r];
      float mx = m_old;
      for (int j = 0; j < 64; ++j) {
        float x = Ss[r * 64 + j];
        if (key0 + j > qa) x = -1e9f;
        mx = fmaxf(mx, x);
      }
      float c = __expf(m_old - mx);
      float sum = 0.0f;
      for (int j = 0; j < 64; ++j) {
        float x = Ss[r * 64 + j];
        if (key0 + j > qa) x = -1e9f;
        float p = __expf(x - mx);
        sum += p;
        Ps[r * 64 + j] = f2bf(p);
      }
      mrow[r] = mx;
      crow[r] = c;
      lrow[r] = lrow[r] * c + sum;
    }
    __syncthreads();

    // ---- rescale accumulators, then O += P * V ----
    {
      const int half = lane >> 4;
      for (int j = 0; j < 2; ++j)
        for (int e = 0; e < 8; ++e)
          oacc[j][e] *= crow[sm * 16 + e + half * 8];
      for (int kk = 0; kk < 2; ++kk) {
        bf16x16 ap = load_frag(Ps, sm * 16, kk * 32, 64, lane);
        for (int j = 0; j < 2; ++j) {
          bf16x16 bv = load_frag(Vts, (dt0 + j) * 16, kk * 32, 64, lane);
          oacc[j] = wmma_bf16(ap, bv, oacc[j]);
        }
      }
    }
    __syncthreads();
  }

  // ---- normalize and write [S][H*128] bf16 for the O-projection GEMM ----
  {
    const int half = lane >> 4, cl = lane & 15;
    for (int j = 0; j < 2; ++j)
      for (int e = 0; e < 8; ++e) {
        int rl  = sm * 16 + e + half * 8;
        int col = (dt0 + j) * 16 + cl;
        float v = oacc[j][e] / lrow[rl];
        O[(size_t)(q0 + rl) * (H * 128) + h * 128 + col] = f2bf(v);
      }
  }
}

// ---------------------------------------------------------------------------
// Host orchestration
// ---------------------------------------------------------------------------
extern "C" void kernel_launch(void* const* d_in, const int* in_sizes, int n_in,
                              void* d_out, int out_size, void* d_ws, size_t ws_size,
                              hipStream_t stream) {
  (void)in_sizes; (void)n_in; (void)out_size; (void)ws_size;
  const float* hidden = (const float*)d_in[0];
  const float* cosT   = (const float*)d_in[1];
  const float* sinT   = (const float*)d_in[2];
  // d_in[3] (dense additive mask) unused: causality applied analytically.
  const float* Wq  = (const float*)d_in[4];
  const float* Wk  = (const float*)d_in[5];
  const float* Wv  = (const float*)d_in[6];
  const float* Wo  = (const float*)d_in[7];
  const float* qnw = (const float*)d_in[8];
  const float* knw = (const float*)d_in[9];
  float* out = (float*)d_out;

  const int S = 4096, HID = 2048, H = 16, KV = 4, D = 128;
  const int KVD = KV * D;  // 512

  // workspace carve-out (256B aligned)
  char* ws = (char*)d_ws;
  size_t off = 0;
  auto alloc = [&](size_t bytes) {
    off = (off + 255) & ~(size_t)255;
    void* p = ws + off;
    off += bytes;
    return p;
  };
  ushort_t* hb  = (ushort_t*)alloc((size_t)S * HID * 2);    // hidden bf16
  ushort_t* wqb = (ushort_t*)alloc((size_t)HID * HID * 2);
  ushort_t* wkb = (ushort_t*)alloc((size_t)KVD * HID * 2);
  ushort_t* wvb = (ushort_t*)alloc((size_t)KVD * HID * 2);
  ushort_t* wob = (ushort_t*)alloc((size_t)HID * HID * 2);
  float*    qf  = (float*)alloc((size_t)S * HID * 4);       // Q f32 [S, H*D]
  float*    kf  = (float*)alloc((size_t)S * KVD * 4);
  float*    vf  = (float*)alloc((size_t)S * KVD * 4);
  ushort_t* qbn = (ushort_t*)alloc((size_t)H  * S * D * 2); // [H][S][D]
  ushort_t* kbn = (ushort_t*)alloc((size_t)KV * S * D * 2);
  ushort_t* vbn = (ushort_t*)alloc((size_t)KV * S * D * 2);
  ushort_t* ab  = (ushort_t*)alloc((size_t)S * HID * 2);    // attn out bf16

  auto cvt = [&](const float* src, ushort_t* dst, size_t n) {
    cvt_bf16_kernel<<<(unsigned)((n + 255) / 256), 256, 0, stream>>>(src, dst, n);
  };
  cvt(hidden, hb,  (size_t)S * HID);
  cvt(Wq,     wqb, (size_t)HID * HID);
  cvt(Wk,     wkb, (size_t)KVD * HID);
  cvt(Wv,     wvb, (size_t)KVD * HID);
  cvt(Wo,     wob, (size_t)HID * HID);

  // QKV projections (y = x @ W^T) on the WMMA pipe
  gemm_xwT_wmma<<<dim3(HID / 64, S / 128), 256, 0, stream>>>(hb, wqb, qf, nullptr, HID, HID);
  gemm_xwT_wmma<<<dim3(KVD / 64, S / 128), 256, 0, stream>>>(hb, wkb, kf, nullptr, KVD, HID);
  gemm_xwT_wmma<<<dim3(KVD / 64, S / 128), 256, 0, stream>>>(hb, wvb, vf, nullptr, KVD, HID);

  // per-head RMSNorm (+RoPE for Q/K), transpose to [head][S][D] bf16
  norm_rope_kernel<<<S * H,  128, 0, stream>>>(qf, qnw,     cosT, sinT, qbn, H,  S, 1);
  norm_rope_kernel<<<S * KV, 128, 0, stream>>>(kf, knw,     cosT, sinT, kbn, KV, S, 1);
  norm_rope_kernel<<<S * KV, 128, 0, stream>>>(vf, nullptr, cosT, sinT, vbn, KV, S, 0);

  // causal GQA flash attention
  flash_attn_kernel<<<dim3(S / 32, H), 256, 0, stream>>>(qbn, kbn, vbn, ab, S, H, KV);

  // output projection -> f32 result
  gemm_xwT_wmma<<<dim3(HID / 64, S / 128), 256, 0, stream>>>(ab, wob, out, nullptr, HID, HID);
}
